// MultiHeadedAttention_31756988187201
// MI455X (gfx1250) — matrix operand order repaired and checked by reading, not compiled, not run on hardware
//
#include <hip/hip_runtime.h>
#include <cstdint>
#include <cstddef>

// MultiHeadedAttention for B=2, S=2048, D=1024, H=16, DK=64 on gfx1250.
// bf16 WMMA (f32 accumulate); flash-attention with double-buffered
// GLOBAL_LOAD_ASYNC_TO_LDS staging of K/V/bias/mask tiles.

namespace {

constexpr int kBt = 2;
constexpr int kS  = 2048;
constexpr int kD  = 1024;
constexpr int kH  = 16;
constexpr int kDK = 64;

typedef __attribute__((ext_vector_type(16))) __bf16 v16bf;
typedef __attribute__((ext_vector_type(8)))  float  v8f;

__device__ __forceinline__ v8f wmma_bf16(v16bf a, v16bf b, v8f c) {
  return __builtin_amdgcn_wmma_f32_16x16x32_bf16(
      /*neg_a=*/false, a, /*neg_b=*/false, b,
      /*c_mod=*/(short)0, c, /*reuse_a=*/false, /*reuse_b=*/false);
}

__device__ __forceinline__ __bf16 f2bf(float f) { return (__bf16)f; }

// Async copy 16B global -> LDS (per lane). VDST VGPR carries the LDS byte
// address (low 32 bits of a generic pointer to a __shared__ object are the
// wave-relative LDS offset). Tracked by ASYNCcnt.
__device__ __forceinline__ void async_copy_b128(void* lds, const void* g) {
  asm volatile("global_load_async_to_lds_b128 %0, %1, off"
               :: "v"((uint32_t)(uintptr_t)lds),
                  "v"((unsigned long long)(uintptr_t)g)
               : "memory");
}

__device__ __forceinline__ void wait_async0() {
#if __has_builtin(__builtin_amdgcn_s_wait_asynccnt)
  __builtin_amdgcn_s_wait_asynccnt(0);
#else
  asm volatile("s_wait_asynccnt 0" ::: "memory");
#endif
}

// A fragment 16x32 bf16. Lane L: m = L&15, half = L>>4.
// VGPR v holds K pair at k = kbase + 16*half + 2*v.
__device__ __forceinline__ v16bf load_a_frag(const __bf16* src, int row, int ld,
                                             int kbase, int half) {
  v16bf a;
#pragma unroll
  for (int v = 0; v < 8; ++v) {
    int k = kbase + (half << 4) + (v << 1);
    a[2 * v]     = src[row * ld + k];
    a[2 * v + 1] = src[row * ld + k + 1];
  }
  return a;
}

// B fragment 32x16 bf16, source stored [n][k] row-major (ld = k-stride).
__device__ __forceinline__ v16bf load_b_frag_nk(const __bf16* src, int n, int ld,
                                                int kbase, int half) {
  v16bf b;
#pragma unroll
  for (int v = 0; v < 8; ++v) {
    int k = kbase + (half << 4) + (v << 1);
    b[2 * v]     = src[n * ld + k];
    b[2 * v + 1] = src[n * ld + k + 1];
  }
  return b;
}

// B fragment 32x16 bf16, source stored [k][n] row-major (ld = n-stride).
__device__ __forceinline__ v16bf load_b_frag_kn(const __bf16* src, int n, int ld,
                                                int kbase, int half) {
  v16bf b;
#pragma unroll
  for (int v = 0; v < 8; ++v) {
    int k = kbase + (half << 4) + (v << 1);
    b[2 * v]     = src[k * ld + n];
    b[2 * v + 1] = src[(k + 1) * ld + n];
  }
  return b;
}

// ---------------------------------------------------------------------------
// GEMM: Out[M=4096, N=1024] = (A[4096,1024] @ W[1024,1024]^T + bias) * scale
// OUT_MODE 0: bf16 [M][N]   1: bf16 head-split [B,H,S,DK]   2: f32 [M][N]
// ---------------------------------------------------------------------------
template <int OUT_MODE, bool A_BF16>
__global__ __launch_bounds__(128) void gemm_wmma_kernel(
    const void* __restrict__ Aptr, const float* __restrict__ W,
    const float* __restrict__ bias, void* __restrict__ Out, float scale) {
  __shared__ __bf16 As[64 * 32];
  __shared__ __bf16 Bs[64 * 32];

  const int tid  = threadIdx.x;
  const int lane = tid & 31;
  const int wv   = tid >> 5;
  const int nloc = lane & 15;
  const int half = lane >> 4;
  const int m0 = blockIdx.x * 64;
  const int n0 = blockIdx.y * 64;

  v8f acc[4];
#pragma unroll
  for (int i = 0; i < 4; ++i)
#pragma unroll
    for (int j = 0; j < 8; ++j) acc[i][j] = 0.f;

  for (int k0 = 0; k0 < kD; k0 += 32) {
    __syncthreads();
    if constexpr (A_BF16) {
      const __bf16* A = (const __bf16*)Aptr;
#pragma unroll
      for (int i = 0; i < 4; ++i) {
        int q = tid + i * 128;
        int r = q >> 3;
        int c0 = (q & 7) << 2;
        *(uint2*)&As[r * 32 + c0] =
            *(const uint2*)&A[(size_t)(m0 + r) * kD + k0 + c0];
      }
    } else {
      const float* A = (const float*)Aptr;
#pragma unroll
      for (int i = 0; i < 4; ++i) {
        int q = tid + i * 128;
        int r = q >> 3;
        int c0 = (q & 7) << 2;
        float4 f = *(const float4*)&A[(size_t)(m0 + r) * kD + k0 + c0];
        As[r * 32 + c0 + 0] = f2bf(f.x);
        As[r * 32 + c0 + 1] = f2bf(f.y);
        As[r * 32 + c0 + 2] = f2bf(f.z);
        As[r * 32 + c0 + 3] = f2bf(f.w);
      }
    }
#pragma unroll
    for (int i = 0; i < 4; ++i) {
      int q = tid + i * 128;
      int r = q >> 3;
      int c0 = (q & 7) << 2;
      float4 f = *(const float4*)&W[(size_t)(n0 + r) * kD + k0 + c0];
      Bs[r * 32 + c0 + 0] = f2bf(f.x);
      Bs[r * 32 + c0 + 1] = f2bf(f.y);
      Bs[r * 32 + c0 + 2] = f2bf(f.z);
      Bs[r * 32 + c0 + 3] = f2bf(f.w);
    }
    __syncthreads();

    v16bf bf = load_b_frag_nk(Bs, wv * 16 + nloc, 32, 0, half);
#pragma unroll
    for (int mt = 0; mt < 4; ++mt) {
      v16bf af = load_a_frag(As, mt * 16 + nloc, 32, 0, half);
      acc[mt] = wmma_bf16(af, bf, acc[mt]);
    }
  }

  const int col = n0 + wv * 16 + nloc;
  const float bcol = bias[col];
#pragma unroll
  for (int mt = 0; mt < 4; ++mt) {
#pragma unroll
    for (int v = 0; v < 8; ++v) {
      const int row = m0 + mt * 16 + v + 8 * half;
      const float val = (acc[mt][v] + bcol) * scale;
      if constexpr (OUT_MODE == 0) {
        ((__bf16*)Out)[(size_t)row * kD + col] = f2bf(val);
      } else if constexpr (OUT_MODE == 1) {
        const int b = row >> 11;
        const int s = row & (kS - 1);
        const int h = col >> 6;
        const int d = col & (kDK - 1);
        ((__bf16*)Out)[(((size_t)(b * kH + h) * kS) + s) * kDK + d] = f2bf(val);
      } else {
        ((float*)Out)[(size_t)row * kD + col] = val;
      }
    }
  }
}

// ---------------------------------------------------------------------------
// Flash attention. Grid (S/64, H, B), block 128 (4 waves).
// Double-buffered async staging: K/V tiles (bf16), rel_pos_bias tile (f32),
// mask tile (u8). One barrier per key tile; tile t+1 streams while tile t
// computes. Q pre-scaled by 1/sqrt(DK).
// ---------------------------------------------------------------------------
__global__ __launch_bounds__(128) void flash_attn_kernel(
    const __bf16* __restrict__ Qp, const __bf16* __restrict__ Kp,
    const __bf16* __restrict__ Vp, const unsigned char* __restrict__ mask,
    const float* __restrict__ relb, __bf16* __restrict__ Ctx) {
  __shared__ __bf16 Qs[64 * 64];
  __shared__ __bf16 Ks[2][64 * 64];
  __shared__ __bf16 Vs[2][64 * 64];
  __shared__ float  BiasS[2][64 * 64];
  __shared__ unsigned char MaskS[2][64 * 64];
  __shared__ __bf16 Ps[4][16 * 64];  // per-wave P bounce (C -> A layout)

  const int tid  = threadIdx.x;
  const int lane = tid & 31;
  const int wv   = tid >> 5;
  const int nloc = lane & 15;
  const int half = lane >> 4;
  const int q0 = blockIdx.x * 64;
  const int h  = blockIdx.y;
  const int b  = blockIdx.z;
  const size_t bh = (size_t)b * kH + h;

  // Issue one key-tile's worth of async LDS loads (18 instr/wave):
  // K: 4, V: 4, bias: 8, mask: 2.
  auto issue_tile = [&](int t, int buf) {
    const __bf16* kbase = Kp + (bh * kS + (size_t)t * 64) * kDK;
    const __bf16* vbase = Vp + (bh * kS + (size_t)t * 64) * kDK;
#pragma unroll
    for (int i = 0; i < 4; ++i) {
      int j = tid + i * 128;  // 512 x 16B covers 64x64 bf16
      async_copy_b128(&Ks[buf][j * 8], kbase + j * 8);
      async_copy_b128(&Vs[buf][j * 8], vbase + j * 8);
    }
#pragma unroll
    for (int i = 0; i < 8; ++i) {
      int j = tid + i * 128;  // 1024 x 16B covers 64x64 f32
      int r = j >> 4;
      int c = (j & 15) << 2;
      async_copy_b128(&BiasS[buf][r * 64 + c],
                      relb + ((size_t)h * kS + q0 + r) * kS + (size_t)t * 64 + c);
    }
#pragma unroll
    for (int i = 0; i < 2; ++i) {
      int j = tid + i * 128;  // 256 x 16B covers 64x64 u8
      int r = j >> 2;
      int c = (j & 3) << 4;
      async_copy_b128(&MaskS[buf][r * 64 + c],
                      mask + ((size_t)b * kS + q0 + r) * kS + (size_t)t * 64 + c);
    }
  };

  // Prologue: Q tile + key tile 0.
#pragma unroll
  for (int i = 0; i < 4; ++i) {
    int j = tid + i * 128;
    async_copy_b128(&Qs[j * 8], Qp + (bh * kS + q0) * kDK + j * 8);
  }
  issue_tile(0, 0);

  v8f oacc[4];
  float mrow[8], lrow[8];
#pragma unroll
  for (int i = 0; i < 4; ++i)
#pragma unroll
    for (int j = 0; j < 8; ++j) oacc[i][j] = 0.f;
#pragma unroll
  for (int v = 0; v < 8; ++v) { mrow[v] = -1e30f; lrow[v] = 0.f; }

  for (int t = 0; t < kS / 64; ++t) {
    const int buf = t & 1;
    wait_async0();       // this wave's loads for tile t landed in LDS
    __syncthreads();     // all waves: tile t ready; prev reads of buf^1 done
    if (t + 1 < kS / 64) issue_tile(t + 1, buf ^ 1);  // overlap with compute

    // S = Q(16x64) @ K^T(64x64): 4 N-tiles, 2 K-steps of 32.
    v8f sacc[4];
#pragma unroll
    for (int i = 0; i < 4; ++i)
#pragma unroll
      for (int j = 0; j < 8; ++j) sacc[i][j] = 0.f;

#pragma unroll
    for (int ks = 0; ks < kDK; ks += 32) {
      v16bf aq = load_a_frag(Qs, wv * 16 + nloc, kDK, ks, half);
#pragma unroll
      for (int nt = 0; nt < 4; ++nt) {
        v16bf bk = load_b_frag_nk(Ks[buf], nt * 16 + nloc, kDK, ks, half);
        sacc[nt] = wmma_bf16(aq, bk, sacc[nt]);
      }
    }

    // Fused bias + mask from LDS tiles.
#pragma unroll
    for (int nt = 0; nt < 4; ++nt) {
      const int kcl = nt * 16 + nloc;
#pragma unroll
      for (int v = 0; v < 8; ++v) {
        const int ql = wv * 16 + v + 8 * half;
        float sv = sacc[nt][v] + BiasS[buf][ql * 64 + kcl];
        if (MaskS[buf][ql * 64 + kcl]) sv = -1e9f;
        sacc[nt][v] = sv;
      }
    }

    // Online softmax. Row m = v + 8*half lives in 16 lanes of one half-wave.
#pragma unroll
    for (int v = 0; v < 8; ++v) {
      float mx = sacc[0][v];
#pragma unroll
      for (int nt = 1; nt < 4; ++nt) mx = fmaxf(mx, sacc[nt][v]);
#pragma unroll
      for (int off = 8; off >= 1; off >>= 1)
        mx = fmaxf(mx, __shfl_xor(mx, off, 16));
      const float mnew  = fmaxf(mrow[v], mx);
      const float alpha = __expf(mrow[v] - mnew);
      float rs = 0.f;
#pragma unroll
      for (int nt = 0; nt < 4; ++nt) {
        const float p = __expf(sacc[nt][v] - mnew);
        sacc[nt][v] = p;
        rs += p;
      }
#pragma unroll
      for (int off = 8; off >= 1; off >>= 1) rs += __shfl_xor(rs, off, 16);
      lrow[v] = lrow[v] * alpha + rs;
      mrow[v] = mnew;
#pragma unroll
      for (int dt = 0; dt < 4; ++dt) oacc[dt][v] *= alpha;
    }

    // Bounce P through per-wave LDS (same-wave LDS ops are in-order; no
    // cross-wave consumers -> no barrier needed).
#pragma unroll
    for (int nt = 0; nt < 4; ++nt)
#pragma unroll
      for (int v = 0; v < 8; ++v)
        Ps[wv][(v + 8 * half) * 64 + nt * 16 + nloc] = f2bf(sacc[nt][v]);

    // O += P(16x64) @ V(64x64).
#pragma unroll
    for (int ks = 0; ks < 64; ks += 32) {
      v16bf ap = load_a_frag(Ps[wv], nloc, 64, ks, half);
#pragma unroll
      for (int dt = 0; dt < 4; ++dt) {
        v16bf bvv = load_b_frag_kn(Vs[buf], dt * 16 + nloc, kDK, ks, half);
        oacc[dt] = wmma_bf16(ap, bvv, oacc[dt]);
      }
    }
  }

  // Normalize and store context bf16 back to [B,S,D] (merge heads).
#pragma unroll
  for (int dt = 0; dt < 4; ++dt) {
#pragma unroll
    for (int v = 0; v < 8; ++v) {
      const float l = lrow[v];
      const float val = (l > 0.f) ? oacc[dt][v] / l : 0.f;
      const int sr  = q0 + wv * 16 + v + 8 * half;
      const int col = h * kDK + dt * 16 + nloc;
      Ctx[((size_t)b * kS + sr) * kD + col] = f2bf(val);
    }
  }
}

}  // namespace

extern "C" void kernel_launch(void* const* d_in, const int* in_sizes, int n_in,
                              void* d_out, int out_size, void* d_ws, size_t ws_size,
                              hipStream_t stream) {
  (void)in_sizes; (void)n_in; (void)out_size; (void)ws_size;

  const float* query = (const float*)d_in[0];
  const float* key   = (const float*)d_in[1];
  const float* value = (const float*)d_in[2];
  const unsigned char* mask = (const unsigned char*)d_in[3];  // jax bool = 1B
  const float* relb  = (const float*)d_in[4];
  const float* Wq = (const float*)d_in[5];
  const float* bq = (const float*)d_in[6];
  const float* Wk = (const float*)d_in[7];
  const float* bk = (const float*)d_in[8];
  const float* Wv = (const float*)d_in[9];
  const float* bv = (const float*)d_in[10];
  const float* Wo = (const float*)d_in[11];
  const float* bo = (const float*)d_in[12];

  const size_t nBSD = (size_t)kBt * kS * kD;
  __bf16* Qp  = (__bf16*)d_ws;
  __bf16* Kp  = Qp + nBSD;
  __bf16* Vp  = Kp + nBSD;
  __bf16* Ctx = Vp + nBSD;

  const dim3 gblk(128);
  const dim3 ggrid((kBt * kS) / 64, kD / 64);  // (64, 16)
  const float qscale = 0.125f;                 // 1/sqrt(DK), folded into Q

  gemm_wmma_kernel<1, false><<<ggrid, gblk, 0, stream>>>(query, Wq, bq, Qp, qscale);
  gemm_wmma_kernel<1, false><<<ggrid, gblk, 0, stream>>>(key,   Wk, bk, Kp, 1.0f);
  gemm_wmma_kernel<1, false><<<ggrid, gblk, 0, stream>>>(value, Wv, bv, Vp, 1.0f);

  const dim3 agrid(kS / 64, kH, kBt);          // (32, 16, 2)
  flash_attn_kernel<<<agrid, gblk, 0, stream>>>(Qp, Kp, Vp, mask, relb, Ctx);

  gemm_wmma_kernel<2, true><<<ggrid, gblk, 0, stream>>>(Ctx, Wo, bo, d_out, 1.0f);
}
